// VisionModel_29738353557814
// MI455X (gfx1250) — matrix operand order repaired and checked by reading, not compiled, not run on hardware
//
#include <hip/hip_runtime.h>
#include <math.h>

// ---------------------------------------------------------------------------
// Types
// ---------------------------------------------------------------------------
typedef __attribute__((ext_vector_type(16))) __bf16          v16bf;
typedef __attribute__((ext_vector_type(8)))  float           v8f;
typedef __attribute__((ext_vector_type(16))) unsigned short  v16us;
typedef __attribute__((ext_vector_type(8)))  unsigned short  us8;

__device__ __forceinline__ unsigned short f2bf(float f) {
  union { float f; unsigned u; } v; v.f = f;
  unsigned u = v.u;
  unsigned r = u + 0x7FFFu + ((u >> 16) & 1u);   // round-to-nearest-even
  return (unsigned short)(r >> 16);
}

__device__ __forceinline__ v8f wmma_bf16(v16us a, v16us b, v8f c) {
  return __builtin_amdgcn_wmma_f32_16x16x32_bf16(
      false, __builtin_bit_cast(v16bf, a),
      false, __builtin_bit_cast(v16bf, b),
      (short)0, c, false, false);
}

__device__ __forceinline__ v16us join8(us8 lo, us8 hi) {
  return __builtin_shufflevector(lo, hi, 0,1,2,3,4,5,6,7,8,9,10,11,12,13,14,15);
}

// ---------------------------------------------------------------------------
// Model constants
// ---------------------------------------------------------------------------
#define NTOK   3328
#define HDIM   1152
#define NHEAD  16
#define HD     72
#define HDP    96       // head dim zero-padded for WMMA K (3 x 32)
#define QKVD   3456
#define INTERD 4304
#define MD     4608
#define OUTD   2048
#define MROWS  832
#define SEG0   2304

// ---------------------------------------------------------------------------
// token -> (row, col) within its grid (matches _pos_ids / merge shuffle)
// ---------------------------------------------------------------------------
__device__ __forceinline__ void tok_rc(int i, int& r, int& c, int& grid) {
  if (i < SEG0) {                     // grid 0: 48x48, 24x24 merge blocks
    int i4 = i & 3, blk = i >> 2;
    int br = blk / 24, bc = blk % 24;
    r = br * 2 + (i4 >> 1); c = bc * 2 + (i4 & 1); grid = 0;
  } else {                            // grid 1: 32x32, 16x16 merge blocks
    int j = i - SEG0;
    int i4 = j & 3, blk = j >> 2;
    int br = blk / 16, bc = blk % 16;
    r = br * 2 + (i4 >> 1); c = bc * 2 + (i4 & 1); grid = 1;
  }
}

// ---------------------------------------------------------------------------
// f32 -> bf16 converter
// ---------------------------------------------------------------------------
__global__ __launch_bounds__(256) void k_f32_to_bf16(
    const float* __restrict__ in, unsigned short* __restrict__ out, long n) {
  long i = (long)blockIdx.x * blockDim.x + threadIdx.x;
  long stride = (long)gridDim.x * blockDim.x;
  for (; i < n; i += stride) out[i] = f2bf(in[i]);
}

// ---------------------------------------------------------------------------
// LayerNorm (per row of D), writes bf16
// ---------------------------------------------------------------------------
__global__ __launch_bounds__(256) void k_layernorm(
    const float* __restrict__ x, const float* __restrict__ s,
    const float* __restrict__ b, unsigned short* __restrict__ out, int D) {
  int row = blockIdx.x, tid = threadIdx.x;
  const float* xr = x + (size_t)row * D;
  __shared__ float r1[256], r2[256];
  float sum = 0.f, sum2 = 0.f;
  for (int j = tid; j < D; j += 256) { float v = xr[j]; sum += v; sum2 += v * v; }
  r1[tid] = sum; r2[tid] = sum2; __syncthreads();
  for (int sft = 128; sft > 0; sft >>= 1) {
    if (tid < sft) { r1[tid] += r1[tid + sft]; r2[tid] += r2[tid + sft]; }
    __syncthreads();
  }
  float mean = r1[0] / (float)D;
  float var  = r2[0] / (float)D - mean * mean;
  float inv  = rsqrtf(var + 1e-6f);
  unsigned short* orow = out + (size_t)row * D;
  for (int j = tid; j < D; j += 256)
    orow[j] = f2bf(s[j] * ((xr[j] - mean) * inv) + b[j]);
}

// ---------------------------------------------------------------------------
// Bilinear position-embedding add (interp_pos with merge shuffle)
// ---------------------------------------------------------------------------
__global__ __launch_bounds__(256) void k_posadd(
    float* __restrict__ h, const float* __restrict__ table) {
  int tok = blockIdx.x;
  int r, c, grid; tok_rc(tok, r, c, grid);
  float hi, wi;
  if (grid == 0) { hi = (float)r; wi = (float)c; }
  else           { float f = 47.f / 31.f; hi = r * f; wi = c * f; }
  int hf = (int)floorf(hi), wf = (int)floorf(wi);
  int hc = min(hf + 1, 47), wc = min(wf + 1, 47);
  float dh = hi - hf, dw = wi - wf;
  float w00 = (1.f - dh) * (1.f - dw), w01 = (1.f - dh) * dw;
  float w10 = dh * (1.f - dw),         w11 = dh * dw;
  const float* t00 = table + (size_t)(hf * 48 + wf) * HDIM;
  const float* t01 = table + (size_t)(hf * 48 + wc) * HDIM;
  const float* t10 = table + (size_t)(hc * 48 + wf) * HDIM;
  const float* t11 = table + (size_t)(hc * 48 + wc) * HDIM;
  float* hr = h + (size_t)tok * HDIM;
  for (int j = threadIdx.x; j < HDIM; j += 256)
    hr[j] += w00 * t00[j] + w01 * t01[j] + w10 * t10[j] + w11 * t11[j];
}

// ---------------------------------------------------------------------------
// RoPE + head-major bf16 repack of q,k,v, zero-padded to HDP per head
// layout: [head][tok][HDP]
// ---------------------------------------------------------------------------
__global__ __launch_bounds__(256) void k_rope(
    const float* __restrict__ qkv, unsigned short* __restrict__ qb,
    unsigned short* __restrict__ kb, unsigned short* __restrict__ vb) {
  int tok = blockIdx.x;
  int r, c, grid; tok_rc(tok, r, c, grid); (void)grid;
  const float* base = qkv + (size_t)tok * QKVD;
  for (int jj = threadIdx.x; jj < NHEAD * HDP; jj += 256) {
    int head = jj / HDP, d = jj % HDP;
    size_t oidx = ((size_t)head * NTOK + tok) * HDP + d;
    if (d >= HD) { qb[oidx] = 0; kb[oidx] = 0; vb[oidx] = 0; continue; }
    int j = head * HD + d;
    float q = base[j], k = base[HDIM + j], v = base[2 * HDIM + j];
    int j36 = d % 36;
    float pos = (j36 < 18) ? (float)r : (float)c;
    int fi = (j36 < 18) ? j36 : (j36 - 18);
    float ang = pos * __powf(10000.f, -(float)fi / 18.f);
    float cs = __cosf(ang), sn = __sinf(ang);
    float qo, ko;
    if (d < 36) {
      float qp = base[j + 36], kp = base[HDIM + j + 36];
      qo = q * cs - qp * sn; ko = k * cs - kp * sn;
    } else {
      float qp = base[j - 36], kp = base[HDIM + j - 36];
      qo = q * cs + qp * sn; ko = k * cs + kp * sn;
    }
    qb[oidx] = f2bf(qo); kb[oidx] = f2bf(ko); vb[oidx] = f2bf(v);
  }
}

// ---------------------------------------------------------------------------
// WMMA GEMM: C = act(A(bf16, MxK) * B(bf16, KxN) + bias) [+ resid]
// block = 256 thr = 8 waves, tile 128x128, each wave 32x64 (2x4 WMMA accs)
// All M,N,K in this model are multiples of 16 -> tails are pure zero-fill.
// ---------------------------------------------------------------------------
enum { F_BIAS = 1, F_GTANH = 2, F_GERF = 4, F_RES = 8, F_BF16 = 16, F_F32 = 32 };
#define LDT 40   // padded LDS row stride (elements)

__global__ __launch_bounds__(256) void k_gemm(
    const unsigned short* __restrict__ A, const unsigned short* __restrict__ B,
    const float* __restrict__ bias, const float* __restrict__ resid,
    float* __restrict__ Cf, unsigned short* __restrict__ Cbf,
    int M, int N, int K, int flags) {
  __shared__ unsigned short sA[128 * LDT];   // [m][k]
  __shared__ unsigned short sB[128 * LDT];   // [n][k] (transposed)

  const int tid = threadIdx.x;
  const int bm = blockIdx.x * 128, bn = blockIdx.y * 128;
  const int wave = tid >> 5, lane = tid & 31;
  const int wm = wave & 3, wn = wave >> 2;        // 4 row-waves x 2 col-waves
  const int lhalf = lane >> 4, l16 = lane & 15;

  v8f acc[2][4] = {};
  const us8 z8 = (us8)(unsigned short)0;

  for (int k0 = 0; k0 < K; k0 += 32) {
    // ---- stage A tile: thread loads 16 bf16 of one row (or zero-fills) ----
    {
      int ra = tid >> 1, ca = (tid & 1) << 4;
      int gr = bm + ra, kc = k0 + ca;
      us8 v0 = z8, v1 = z8;
      if (gr < M && kc < K) {           // K % 16 == 0 -> whole 16 in range
        const us8* p = (const us8*)(A + (size_t)gr * K + kc);
        v0 = p[0]; v1 = p[1];
      }
      *(us8*)&sA[ra * LDT + ca]     = v0;
      *(us8*)&sA[ra * LDT + ca + 8] = v1;
    }
    // ---- stage B tile transposed (or zero-fill) ----
    {
      int kr = tid >> 3, nc = (tid & 7) << 4;
      int gk = k0 + kr, gn = bn + nc;
      us8 b0 = z8, b1 = z8;
      if (gk < K && gn < N) {           // N % 16 == 0 -> whole 16 in range
        const us8* p = (const us8*)(B + (size_t)gk * N + gn);
        b0 = p[0]; b1 = p[1];
      }
      #pragma unroll
      for (int e = 0; e < 8; ++e) {
        sB[(nc + e) * LDT + kr]     = b0[e];
        sB[(nc + 8 + e) * LDT + kr] = b1[e];
      }
    }
    if (k0 + 32 < K) {   // CDNA5 global_prefetch_b8 for next tiles
      __builtin_prefetch(A + (size_t)(bm + (tid >> 1)) * K + k0 + 32, 0, 1);
      __builtin_prefetch(B + (size_t)(k0 + 32 + (tid >> 3)) * N + bn + ((tid & 7) << 4), 0, 1);
    }
    __syncthreads();

    // ---- fragments + WMMA ----
    v16us aF[2], bF[4];
    #pragma unroll
    for (int mi = 0; mi < 2; ++mi) {
      int row = wm * 32 + mi * 16 + l16;
      int kbase = lhalf * 8;
      us8 lo = *(const us8*)&sA[row * LDT + kbase];
      us8 hi = *(const us8*)&sA[row * LDT + kbase + 16];
      aF[mi] = join8(lo, hi);
    }
    #pragma unroll
    for (int ni = 0; ni < 4; ++ni) {
      int col = wn * 64 + ni * 16 + l16;
      int kb = lhalf * 8;
      us8 lo = *(const us8*)&sB[col * LDT + kb];
      us8 hi = *(const us8*)&sB[col * LDT + kb + 16];
      bF[ni] = join8(lo, hi);
    }
    #pragma unroll
    for (int mi = 0; mi < 2; ++mi)
      #pragma unroll
      for (int ni = 0; ni < 4; ++ni)
        acc[mi][ni] = wmma_bf16(aF[mi], bF[ni], acc[mi][ni]);
    __syncthreads();
  }

  // ---- epilogue: tile-uniform bounds checks (M,N % 16 == 0) ----
  #pragma unroll
  for (int mi = 0; mi < 2; ++mi) {
    int tr = bm + wm * 32 + mi * 16;
    if (tr >= M) continue;
    #pragma unroll
    for (int ni = 0; ni < 4; ++ni) {
      int tc = bn + wn * 64 + ni * 16;
      if (tc >= N) continue;
      int gr0 = tr + lhalf * 8;
      int gc  = tc + l16;
      #pragma unroll
      for (int v = 0; v < 8; ++v) {
        size_t idx = (size_t)(gr0 + v) * N + gc;
        float x = acc[mi][ni][v];
        if (flags & F_BIAS) x += bias[gc];
        if (flags & F_GTANH) {
          float t = 0.7978845608028654f * (x + 0.044715f * x * x * x);
          x = 0.5f * x * (1.f + tanhf(t));
        }
        if (flags & F_GERF) x = 0.5f * x * (1.f + erff(x * 0.70710678118f));
        if (flags & F_RES)  x += resid[idx];
        if (flags & F_F32)  Cf[idx]  = x;
        if (flags & F_BF16) Cbf[idx] = f2bf(x);
      }
    }
  }
}

// ---------------------------------------------------------------------------
// Flash attention: grid(26 qtiles, 16 heads), 8 waves x 16 query rows.
// q/k/v are pre-padded to HDP=96 with zeros -> branch-free fragments.
// ---------------------------------------------------------------------------
__device__ __forceinline__ float redmax16(float x) {
  #pragma unroll
  for (int m = 1; m < 16; m <<= 1) x = fmaxf(x, __shfl_xor(x, m, 32));
  return x;
}
__device__ __forceinline__ float redsum16(float x) {
  #pragma unroll
  for (int m = 1; m < 16; m <<= 1) x += __shfl_xor(x, m, 32);
  return x;
}

__global__ __launch_bounds__(256) void k_attn(
    const unsigned short* __restrict__ qb, const unsigned short* __restrict__ kb,
    const unsigned short* __restrict__ vb, float* __restrict__ o) {
  __shared__ unsigned short sP[8][16 * LDT];   // per wave: 16 rows x 32 probs

  const int head = blockIdx.y;
  const int qt = blockIdx.x;
  int seg_s, seg_e, qbase;
  if (qt < 18) { seg_s = 0;    seg_e = SEG0; qbase = qt * 128; }
  else         { seg_s = SEG0; seg_e = NTOK; qbase = SEG0 + (qt - 18) * 128; }

  const int wave = threadIdx.x >> 5, lane = threadIdx.x & 31;
  const int lhalf = lane >> 4, l16 = lane & 15;
  const int qrow0 = qbase + wave * 16;
  const float scale = 0.1178511302f;   // 1/sqrt(72)

  // Q fragments (16 rows x K=96): vector loads, padding already in memory
  v16us aQ[3];
  {
    const unsigned short* qp = qb + ((size_t)head * NTOK + qrow0 + l16) * HDP;
    #pragma unroll
    for (int cch = 0; cch < 3; ++cch) {
      us8 lo = *(const us8*)(qp + cch * 32 + lhalf * 8);
      us8 hi = *(const us8*)(qp + cch * 32 + lhalf * 8 + 16);
      aQ[cch] = join8(lo, hi);
    }
  }

  float mrun[8], lrun[8];
  #pragma unroll
  for (int v = 0; v < 8; ++v) { mrun[v] = -1e30f; lrun[v] = 0.f; }
  v8f oacc[5] = {};

  for (int kk = seg_s; kk < seg_e; kk += 32) {
    // S tiles for two 16-key sub-tiles
    v8f sv[2];
    #pragma unroll
    for (int sub = 0; sub < 2; ++sub) {
      const unsigned short* kp =
          kb + ((size_t)head * NTOK + kk + sub * 16 + l16) * HDP;
      v8f s = {};
      #pragma unroll
      for (int cch = 0; cch < 3; ++cch) {
        const us8* p = (const us8*)(kp + cch * 32 + lhalf * 8);
        v16us bK = join8(p[0], p[2]);
        s = wmma_bf16(aQ[cch], bK, s);
      }
      sv[sub] = s;
    }

    // online softmax (rows live in 16-lane halves)
    #pragma unroll
    for (int v = 0; v < 8; ++v) {
      float s0 = sv[0][v] * scale, s1 = sv[1][v] * scale;
      float rm = redmax16(fmaxf(s0, s1));
      float mn = fmaxf(mrun[v], rm);
      float corr = __expf(mrun[v] - mn);
      mrun[v] = mn;
      float p0 = __expf(s0 - mn), p1 = __expf(s1 - mn);
      float rs = redsum16(p0 + p1);
      lrun[v] = lrun[v] * corr + rs;
      #pragma unroll
      for (int t = 0; t < 5; ++t) oacc[t][v] *= corr;
      int prow = lhalf * 8 + v;
      sP[wave][prow * LDT + l16]      = f2bf(p0);
      sP[wave][prow * LDT + 16 + l16] = f2bf(p1);
    }
    asm volatile("s_wait_dscnt 0" ::: "memory");   // same-wave LDS RAW fence

    // P as A-fragment (16x32)
    v16us aP;
    {
      us8 lo = *(const us8*)&sP[wave][l16 * LDT + lhalf * 16];
      us8 hi = *(const us8*)&sP[wave][l16 * LDT + lhalf * 16 + 8];
      aP = join8(lo, hi);
    }
    // PV across 5 head-dim tiles (72 padded to 80 < HDP, pad cols are zeros)
    #pragma unroll
    for (int t = 0; t < 5; ++t) {
      const unsigned short* vp =
          vb + ((size_t)head * NTOK + kk + lhalf * 16) * HDP + t * 16 + l16;
      v16us bV;
      #pragma unroll
      for (int e = 0; e < 16; ++e) bV[e] = vp[(size_t)e * HDP];
      oacc[t] = wmma_bf16(aP, bV, oacc[t]);
    }
  }

  // write normalized output (token-major, [tok][head*72+d])
  #pragma unroll
  for (int t = 0; t < 5; ++t)
    #pragma unroll
    for (int v = 0; v < 8; ++v) {
      int dim = t * 16 + l16;
      int row = qrow0 + lhalf * 8 + v;
      if (dim < HD)
        o[(size_t)row * HDIM + head * HD + dim] = oacc[t][v] / lrun[v];
    }
}

// ---------------------------------------------------------------------------
// Host helpers
// ---------------------------------------------------------------------------
static void conv_bf16(const float* in, unsigned short* out, long n, hipStream_t s) {
  long blocks = (n + 4095) / 4096; if (blocks > 16384) blocks = 16384;
  k_f32_to_bf16<<<dim3((unsigned)blocks), dim3(256), 0, s>>>(in, out, n);
}

static void run_gemm(const unsigned short* A, const float* Wf32, unsigned short* wtb,
                     const float* bias, const float* resid, float* Cf,
                     unsigned short* Cbf, int M, int N, int K, int flags,
                     hipStream_t s) {
  conv_bf16(Wf32, wtb, (long)K * N, s);
  dim3 g((M + 127) / 128, (N + 127) / 128);
  k_gemm<<<g, dim3(256), 0, s>>>(A, wtb, bias, resid, Cf, Cbf, M, N, K, flags);
}

// ---------------------------------------------------------------------------
// kernel_launch
// ---------------------------------------------------------------------------
extern "C" void kernel_launch(void* const* d_in, const int* in_sizes, int n_in,
                              void* d_out, int out_size, void* d_ws, size_t ws_size,
                              hipStream_t stream) {
  (void)in_sizes; (void)n_in; (void)out_size; (void)ws_size;

  const float* pixels    = (const float*)d_in[0];
  const float* patch_w   = (const float*)d_in[1];
  const float* patch_b   = (const float*)d_in[2];
  const float* pos_table = (const float*)d_in[3];
  auto blkp = [&](int b, int j) { return (const float*)d_in[4 + b * 12 + j]; };
  const int MERG = 52, DSM = 58;   // merger / ds_merger param bases
  float* out = (float*)d_out;

  // workspace carve-up
  size_t off = 0;
  auto take = [&](size_t bytes) {
    size_t o = off; off += (bytes + 255) & ~(size_t)255; return o;
  };
  char* ws = (char*)d_ws;
  float* h             = (float*)(ws + take((size_t)NTOK * HDIM * 4));
  float* qkvf          = (float*)(ws + take((size_t)NTOK * QKVD * 4));
  float* obuf          = (float*)(ws + take((size_t)NTOK * HDIM * 4));
  unsigned short* bfA  = (unsigned short*)(ws + take((size_t)NTOK * 1536 * 2));
  unsigned short* bfB  = (unsigned short*)(ws + take((size_t)NTOK * INTERD * 2));
  unsigned short* qbuf = (unsigned short*)(ws + take((size_t)NHEAD * NTOK * HDP * 2));
  unsigned short* kbuf = (unsigned short*)(ws + take((size_t)NHEAD * NTOK * HDP * 2));
  unsigned short* vbuf = (unsigned short*)(ws + take((size_t)NHEAD * NTOK * HDP * 2));
  unsigned short* wtb  = (unsigned short*)(ws + take((size_t)MD * MD * 2));

  // 1) patch embedding + positional embedding
  conv_bf16(pixels, bfA, (long)NTOK * 1536, stream);
  run_gemm(bfA, patch_w, wtb, patch_b, nullptr, h, nullptr,
           NTOK, HDIM, 1536, F_BIAS | F_F32, stream);
  k_posadd<<<dim3(NTOK), dim3(256), 0, stream>>>(h, pos_table);

  // 2) transformer blocks
  for (int b = 0; b < 4; ++b) {
    k_layernorm<<<dim3(NTOK), dim3(256), 0, stream>>>(
        h, blkp(b, 0), blkp(b, 1), bfA, HDIM);
    run_gemm(bfA, blkp(b, 4), wtb, blkp(b, 5), nullptr, qkvf, nullptr,
             NTOK, QKVD, HDIM, F_BIAS | F_F32, stream);
    k_rope<<<dim3(NTOK), dim3(256), 0, stream>>>(qkvf, qbuf, kbuf, vbuf);
    k_attn<<<dim3(26, NHEAD), dim3(256), 0, stream>>>(qbuf, kbuf, vbuf, obuf);
    conv_bf16(obuf, bfA, (long)NTOK * HDIM, stream);
    run_gemm(bfA, blkp(b, 6), wtb, blkp(b, 7), h, h, nullptr,
             NTOK, HDIM, HDIM, F_BIAS | F_RES | F_F32, stream);
    k_layernorm<<<dim3(NTOK), dim3(256), 0, stream>>>(
        h, blkp(b, 2), blkp(b, 3), bfA, HDIM);
    run_gemm(bfA, blkp(b, 8), wtb, blkp(b, 9), nullptr, nullptr, bfB,
             NTOK, INTERD, HDIM, F_BIAS | F_GTANH | F_BF16, stream);
    run_gemm(bfB, blkp(b, 10), wtb, blkp(b, 11), h, h, nullptr,
             NTOK, HDIM, INTERD, F_BIAS | F_RES | F_F32, stream);

    if (b == 1) {  // deep-supervision merger (postshuffle: LN over 4608)
      k_layernorm<<<dim3(MROWS), dim3(256), 0, stream>>>(
          h, (const float*)d_in[DSM + 0], (const float*)d_in[DSM + 1], bfA, MD);
      run_gemm(bfA, (const float*)d_in[DSM + 2], wtb, (const float*)d_in[DSM + 3],
               nullptr, nullptr, bfB, MROWS, MD, MD, F_BIAS | F_GERF | F_BF16, stream);
      run_gemm(bfB, (const float*)d_in[DSM + 4], wtb, (const float*)d_in[DSM + 5],
               nullptr, out + (size_t)MROWS * OUTD, nullptr,
               MROWS, OUTD, MD, F_BIAS | F_F32, stream);
    }
  }

  // 3) final merger (LN over H=1152, then contiguous reshape to 832x4608)
  k_layernorm<<<dim3(NTOK), dim3(256), 0, stream>>>(
      h, (const float*)d_in[MERG + 0], (const float*)d_in[MERG + 1], bfA, HDIM);
  run_gemm(bfA, (const float*)d_in[MERG + 2], wtb, (const float*)d_in[MERG + 3],
           nullptr, nullptr, bfB, MROWS, MD, MD, F_BIAS | F_GERF | F_BF16, stream);
  run_gemm(bfB, (const float*)d_in[MERG + 4], wtb, (const float*)d_in[MERG + 5],
           nullptr, out, nullptr, MROWS, OUTD, MD, F_BIAS | F_F32, stream);
}
